// TransformerConvNet_88553635709219
// MI455X (gfx1250) — matrix operand order repaired and checked by reading, not compiled, not run on hardware
//
#include <hip/hip_runtime.h>
#include <hip/hip_bf16.h>
#include <math.h>

typedef __attribute__((ext_vector_type(16))) _Float16 v16h;
typedef __attribute__((ext_vector_type(8)))  float    v8f;

#define F_IN 64

// ---------------------------------------------------------------------------
// WMMA GEMM:  Y[nRows, COUT] = act( X[nRows, CIN] @ W[CIN, COUT] + bias )
// 256 threads = 8 waves per block; each wave computes TPW consecutive
// 16-row tiles. W is staged in LDS pre-swizzled into B-fragment order so
// each lane fetches its whole fragment as one contiguous 32-byte v16h.
// ---------------------------------------------------------------------------
template<int CIN, int COUT, int ACT>
__launch_bounds__(256)
__global__ void gemm_wmma_kernel(const float* __restrict__ X,
                                 const float* __restrict__ W,
                                 const float* __restrict__ bias,
                                 float* __restrict__ Y, int nRows) {
    constexpr int COUT16 = (COUT + 15) & ~15;
    constexpr int NT  = COUT16 / 16;   // 16-col tiles
    constexpr int KT  = CIN / 32;      // 32-deep K tiles
    constexpr int TPW = 4;             // row-tiles per wave

    // LDS: B fragments, layout [kt][t][lane][16 halves] (32B per lane-frag)
    __shared__ __align__(32) _Float16 Wl[KT * NT * 32 * 16];

    // ---- stage W pre-swizzled: one v16h fragment per (kt, t, lane) group
    for (int g = threadIdx.x; g < KT * NT * 32; g += blockDim.x) {
        const int kt = g / (NT * 32);
        const int t  = (g / 32) % NT;
        const int L  = g & 31;
        const int n  = t * 16 + (L & 15);
        const int kb = kt * 32 + 8 * (L >> 4);
        v16h frag;
#pragma unroll
        for (int i = 0; i < 16; ++i)
            frag[i] = (n < COUT) ? (_Float16)W[(kb + i + (i >= 8 ? 8 : 0)) * COUT + n]
                                 : (_Float16)0.0f;
        *(v16h*)&Wl[g * 16] = frag;
    }
    __syncthreads();

    const int lane   = threadIdx.x & 31;
    const int waveId = threadIdx.x >> 5;
    const int m  = lane & 15;            // A-row / B-col / D-col in tile
    const int hh = lane >> 4;            // lane half
    const int numTiles = nRows >> 4;     // nRows % 16 == 0
    const int tile0 = (blockIdx.x * (blockDim.x >> 5) + waveId) * TPW;

    for (int r = 0; r < TPW; ++r) {
        const int tile = tile0 + r;
        if (tile >= numTiles) return;    // wave-uniform: EXEC stays full

        const float* xrow = X + (size_t)(tile * 16 + m) * CIN;

        v8f acc[NT];
#pragma unroll
        for (int t = 0; t < NT; ++t)
            acc[t] = (v8f){0.f,0.f,0.f,0.f,0.f,0.f,0.f,0.f};

#pragma unroll
        for (int kt = 0; kt < KT; ++kt) {
            const int k0 = kt * 32;
            // A fragment: vectorized 128-bit global loads, cvt to f16
            const float* p0 = xrow + k0 + 8 * hh;        // K = k0+8h .. +7
            const float* p1 = xrow + k0 + 16 + 8 * hh;   // K = k0+16+8h .. +7
            float4 A0 = *(const float4*)p0;
            float4 A1 = *(const float4*)(p0 + 4);
            float4 A2 = *(const float4*)p1;
            float4 A3 = *(const float4*)(p1 + 4);
            v16h a;
            a[0]=(_Float16)A0.x;  a[1]=(_Float16)A0.y;  a[2]=(_Float16)A0.z;  a[3]=(_Float16)A0.w;
            a[4]=(_Float16)A1.x;  a[5]=(_Float16)A1.y;  a[6]=(_Float16)A1.z;  a[7]=(_Float16)A1.w;
            a[8]=(_Float16)A2.x;  a[9]=(_Float16)A2.y;  a[10]=(_Float16)A2.z; a[11]=(_Float16)A2.w;
            a[12]=(_Float16)A3.x; a[13]=(_Float16)A3.y; a[14]=(_Float16)A3.z; a[15]=(_Float16)A3.w;
#pragma unroll
            for (int t = 0; t < NT; ++t) {
                v16h b = *(const v16h*)&Wl[((kt * NT + t) * 32 + lane) * 16];
                acc[t] = __builtin_amdgcn_wmma_f32_16x16x32_f16(
                    false, a, false, b, (short)0, acc[t], false, false);
            }
        }

#pragma unroll
        for (int t = 0; t < NT; ++t) {
            int col = t * 16 + m;
            if (col < COUT) {
                float bv = bias[col];
#pragma unroll
                for (int j = 0; j < 8; ++j) {
                    float val = acc[t][j] + bv;
                    if (ACT == 1) val = val > 0.f ? val : expm1f(val);   // ELU
                    Y[(size_t)(tile * 16 + 8 * hh + j) * COUT + col] = val;
                }
            }
        }
    }
}

// ---------------------------------------------------------------------------
// Edge stage: one wave32 per edge, lane = channel (C/32 channels per lane).
// ---------------------------------------------------------------------------
__device__ __forceinline__ void atomicMaxFloat(float* addr, float val) {
    if (val >= 0.0f) atomicMax((int*)addr, __float_as_int(val));
    else             atomicMin((unsigned int*)addr, __float_as_uint(val));
}

template<int C>
__launch_bounds__(256)
__global__ void edge_alpha_kernel(const float* __restrict__ q,
                                  const float* __restrict__ k,
                                  const float* __restrict__ ea,
                                  const float* __restrict__ We,
                                  const float* __restrict__ be,
                                  const int* __restrict__ src,
                                  const int* __restrict__ dst,
                                  float* __restrict__ alpha,
                                  float* __restrict__ mmax, int nEdges) {
    const int e = blockIdx.x * (blockDim.x >> 5) + (threadIdx.x >> 5);
    if (e >= nEdges) return;
    const int lane = threadIdx.x & 31;
    const int s = src[e], d = dst[e];
    const float eaw = ea[e];
    float partial = 0.f;
#pragma unroll
    for (int cc = 0; cc < C; cc += 32) {
        int c = cc + lane;
        float ke = k[(size_t)s * C + c] + eaw * We[c] + be[c];
        partial += q[(size_t)d * C + c] * ke;
    }
#pragma unroll
    for (int off = 16; off > 0; off >>= 1)
        partial += __shfl_xor(partial, off, 32);
    const float scale = (C == 32) ? 0.1767766952966369f : 0.125f; // 1/sqrt(C)
    float al = partial * scale;
    if (lane == 0) { alpha[e] = al; atomicMaxFloat(&mmax[d], al); }
}

template<int C>
__launch_bounds__(256)
__global__ void edge_accum_kernel(const float* __restrict__ alpha,
                                  const float* __restrict__ mmax,
                                  const float* __restrict__ v,
                                  const float* __restrict__ ea,
                                  const float* __restrict__ We,
                                  const float* __restrict__ be,
                                  const int* __restrict__ src,
                                  const int* __restrict__ dst,
                                  float* __restrict__ acc,
                                  float* __restrict__ snorm, int nEdges) {
    const int e = blockIdx.x * (blockDim.x >> 5) + (threadIdx.x >> 5);
    if (e >= nEdges) return;
    const int lane = threadIdx.x & 31;
    const int s = src[e], d = dst[e];
    const float eaw = ea[e];
    const float a = expf(alpha[e] - mmax[d]);
#pragma unroll
    for (int cc = 0; cc < C; cc += 32) {
        int c = cc + lane;
        float ve = v[(size_t)s * C + c] + eaw * We[c] + be[c];
        atomicAdd(&acc[(size_t)d * C + c], a * ve);
    }
    if (lane == 0) atomicAdd(&snorm[d], a);
}

// acc = 0, snorm = 0, mmax = -inf
__global__ void init_conv_ws(float* acc, int accN, float* snorm, float* mmax,
                             int nNodes) {
    int i = blockIdx.x * blockDim.x + threadIdx.x;
    if (i < accN) acc[i] = 0.f;
    if (i < nNodes) { snorm[i] = 0.f; mmax[i] = -INFINITY; }
}

// out = elu( acc / (s + 1e-16) + skip )
template<int C>
__global__ void finalize_conv(const float* __restrict__ acc,
                              const float* __restrict__ snorm,
                              const float* __restrict__ skip,
                              float* __restrict__ out, int nNodes) {
    int i = blockIdx.x * blockDim.x + threadIdx.x;
    if (i >= nNodes * C) return;
    int n = i / C;
    float val = acc[i] / (snorm[n] + 1e-16f) + skip[i];
    out[i] = val > 0.f ? val : expm1f(val);
}

__global__ void log_softmax_kernel(const float* __restrict__ logits,
                                   float* __restrict__ out, int nRows) {
    int r = blockIdx.x * blockDim.x + threadIdx.x;
    if (r >= nRows) return;
    const float* p = logits + (size_t)r * 10;
    float mx = p[0];
#pragma unroll
    for (int i = 1; i < 10; ++i) mx = fmaxf(mx, p[i]);
    float s = 0.f;
#pragma unroll
    for (int i = 0; i < 10; ++i) s += expf(p[i] - mx);
    float ls = logf(s) + mx;
#pragma unroll
    for (int i = 0; i < 10; ++i) out[(size_t)r * 10 + i] = p[i] - ls;
}

// ---------------------------------------------------------------------------
extern "C" void kernel_launch(void* const* d_in, const int* in_sizes, int n_in,
                              void* d_out, int out_size, void* d_ws, size_t ws_size,
                              hipStream_t stream) {
    const int N = in_sizes[0] / F_IN;       // 100000
    const int E = in_sizes[1] / 2;          // 1600000

    const float* x   = (const float*)d_in[0];
    const int*   ei  = (const int*)d_in[1];
    const int*   src = ei;
    const int*   dst = ei + E;
    const float* ea  = (const float*)d_in[2];
    const float *Wq1=(const float*)d_in[3],  *bq1=(const float*)d_in[4];
    const float *Wk1=(const float*)d_in[5],  *bk1=(const float*)d_in[6];
    const float *Wv1=(const float*)d_in[7],  *bv1=(const float*)d_in[8];
    const float *We1=(const float*)d_in[9],  *be1=(const float*)d_in[10];
    const float *Ws1=(const float*)d_in[11], *bs1=(const float*)d_in[12];
    const float *Wq2=(const float*)d_in[13], *bq2=(const float*)d_in[14];
    const float *Wk2=(const float*)d_in[15], *bk2=(const float*)d_in[16];
    const float *Wv2=(const float*)d_in[17], *bv2=(const float*)d_in[18];
    const float *We2=(const float*)d_in[19], *be2=(const float*)d_in[20];
    const float *Ws2=(const float*)d_in[21], *bs2=(const float*)d_in[22];
    const float *Wf1=(const float*)d_in[23], *bf1=(const float*)d_in[24];
    const float *Wf2=(const float*)d_in[25], *bf2=(const float*)d_in[26];

    float* ws = (float*)d_ws;
    const size_t NB = (size_t)N * 64;
    float* q      = ws;                 // N*64
    float* k      = ws + NB;            // N*64
    float* v      = ws + 2 * NB;        // N*64
    float* skip   = ws + 3 * NB;        // N*64
    float* acc    = ws + 4 * NB;        // N*64
    float* h      = ws + 5 * NB;        // N*64 (h1 then h2)
    float* mmax   = ws + 6 * NB;        // N
    float* snorm  = ws + 6 * NB + N;    // N
    float* alpha  = ws + 6 * NB + 2*N;  // E
    float* hfc    = q;                  // N*128 (reuses q+k after convs)
    float* logits = v;                  // N*10  (reuses v)

    const int numTiles   = N / 16;                       // 6250
    const int gemmBlocks = (numTiles + 8 * 4 - 1) / (8 * 4);  // TPW=4
    const int edgeBlocks = (E + 7) / 8;
    dim3 blk(256);

    // ---------------- conv1: 64 -> 32 ----------------
    gemm_wmma_kernel<64,32,0><<<gemmBlocks, blk, 0, stream>>>(x, Wq1, bq1, q, N);
    gemm_wmma_kernel<64,32,0><<<gemmBlocks, blk, 0, stream>>>(x, Wk1, bk1, k, N);
    gemm_wmma_kernel<64,32,0><<<gemmBlocks, blk, 0, stream>>>(x, Wv1, bv1, v, N);
    gemm_wmma_kernel<64,32,0><<<gemmBlocks, blk, 0, stream>>>(x, Ws1, bs1, skip, N);
    init_conv_ws<<<(N * 32 + 255) / 256, blk, 0, stream>>>(acc, N * 32, snorm, mmax, N);
    edge_alpha_kernel<32><<<edgeBlocks, blk, 0, stream>>>(q, k, ea, We1, be1, src, dst, alpha, mmax, E);
    edge_accum_kernel<32><<<edgeBlocks, blk, 0, stream>>>(alpha, mmax, v, ea, We1, be1, src, dst, acc, snorm, E);
    finalize_conv<32><<<(N * 32 + 255) / 256, blk, 0, stream>>>(acc, snorm, skip, h, N);

    // ---------------- conv2: 32 -> 64 ----------------
    gemm_wmma_kernel<32,64,0><<<gemmBlocks, blk, 0, stream>>>(h, Wq2, bq2, q, N);
    gemm_wmma_kernel<32,64,0><<<gemmBlocks, blk, 0, stream>>>(h, Wk2, bk2, k, N);
    gemm_wmma_kernel<32,64,0><<<gemmBlocks, blk, 0, stream>>>(h, Wv2, bv2, v, N);
    gemm_wmma_kernel<32,64,0><<<gemmBlocks, blk, 0, stream>>>(h, Ws2, bs2, skip, N);
    init_conv_ws<<<(N * 64 + 255) / 256, blk, 0, stream>>>(acc, N * 64, snorm, mmax, N);
    edge_alpha_kernel<64><<<edgeBlocks, blk, 0, stream>>>(q, k, ea, We2, be2, src, dst, alpha, mmax, E);
    edge_accum_kernel<64><<<edgeBlocks, blk, 0, stream>>>(alpha, mmax, v, ea, We2, be2, src, dst, acc, snorm, E);
    finalize_conv<64><<<(N * 64 + 255) / 256, blk, 0, stream>>>(acc, snorm, skip, h, N);

    // ---------------- MLP head ----------------
    gemm_wmma_kernel<64,128,1><<<gemmBlocks, blk, 0, stream>>>(h, Wf1, bf1, hfc, N);
    gemm_wmma_kernel<128,10,0><<<gemmBlocks, blk, 0, stream>>>(hfc, Wf2, bf2, logits, N);
    log_softmax_kernel<<<(N + 255) / 256, blk, 0, stream>>>(logits, (float*)d_out, N);
}
